// unit_gcn_3032246911629
// MI455X (gfx1250) — matrix operand order repaired and checked by reading, not compiled, not run on hardware
//
#include <hip/hip_runtime.h>
#include <hip/hip_bf16.h>

typedef __bf16 v16bf __attribute__((ext_vector_type(16)));
typedef __bf16 v8bf  __attribute__((ext_vector_type(8)));
typedef float  v8f   __attribute__((ext_vector_type(8)));
typedef float  v4f   __attribute__((ext_vector_type(4)));
typedef unsigned int u32x4 __attribute__((ext_vector_type(4)));
typedef int i32x4 __attribute__((ext_vector_type(4)));
typedef int i32x8 __attribute__((ext_vector_type(8)));

#define NB 64
#define CB 64
#define TB 300
#define VB 25
#define ROWS 1600              // (o,w) = 64*25
#define COLS 19200             // (n,t) = 64*300
#define KDIM 1600              // (c,v) = 64*25
#define KSTEPS 50              // KDIM / 32 (even -> clean unroll-by-2)
#define NTV  480000            // N*T*V per-channel count for BN
#define WL_STRIDE 40           // bf16 elements per LDS row (80B; 64B data + 16B pad)

// ---------------- zero BN stats accumulator (must be re-zeroed every launch) --
__global__ void zero_stats_kernel(float* stats) {
    stats[threadIdx.x] = 0.0f;   // [0:64) sum, [64:128) sumsq, [128:192) scale, [192:256) shift
}

// ---------------- BigW[(o,w),(c,v)] = sum_s Wc[s,o,c] * PA[s,v,w]/(||PA[s,:,w]||+1e-4)
__global__ void bigw_kernel(const float* __restrict__ PA, const float* __restrict__ Wc,
                            __bf16* __restrict__ BigW) {
    int idx = blockIdx.x * 256 + threadIdx.x;     // 2,560,000 = 10000*256 exact
    int row = idx / KDIM, k = idx % KDIM;
    int o = row / VB, w = row % VB;
    int c = k / VB,  v = k % VB;
    float acc = 0.0f;
    #pragma unroll
    for (int s = 0; s < 3; ++s) {
        float nrm = 0.0f;
        #pragma unroll
        for (int vv = 0; vv < VB; ++vv) {
            float a = PA[s * 625 + vv * VB + w];
            nrm += a * a;
        }
        float Anorm = PA[s * 625 + v * VB + w] / (sqrtf(nrm) + 1e-4f);
        acc += Wc[s * 4096 + o * CB + c] * Anorm;
    }
    BigW[idx] = (__bf16)acc;
}

// ---------------- Xbf[col][k] = x[n,c,t,v]   (col=(n,t), k=(c,v)), bf16, K-contiguous
__global__ void xprep_kernel(const float* __restrict__ x, __bf16* __restrict__ Xbf) {
    long long idx = (long long)blockIdx.x * 256 + threadIdx.x; // 30,720,000 = 120000*256 exact
    int col = (int)(idx / KDIM), k = (int)(idx % KDIM);
    int n = col / TB, t = col % TB;
    int c = k / VB,  v = k % VB;
    Xbf[idx] = (__bf16)x[(((long long)n * CB + c) * TB + t) * VB + v];
}

// ---------------- Tensor Data Mover: 2D tile (tile_k contiguous bf16 x tile_rows),
// row stride = stride_elems, into LDS with 16B pad every 64B -> LDS row = WL_STRIDE bf16.
// D# encoding per CDNA5 ISA.
__device__ __forceinline__ void tdm_load_2d(unsigned lds_byte_off, const __bf16* gptr,
                                            int tile_k, int tile_rows, long long stride_elems) {
    unsigned long long ga = (unsigned long long)(uintptr_t)gptr;
    u32x4 g0;
    g0[0] = 1u;                                               // count=1, user descriptor
    g0[1] = lds_byte_off;                                     // lds_addr
    g0[2] = (unsigned)(ga & 0xFFFFFFFFu);                     // global_addr[31:0]
    g0[3] = (unsigned)((ga >> 32) & 0x1FFFFFFu) | (2u << 30); // global_addr[56:32] | type=2
    i32x8 g1;
    g1[0] = (1 << 16)      // data_size = 1 -> 2 bytes (bf16)
          | (1 << 20)      // pad_enable
          | (3 << 22)      // pad_interval: code 3 -> every 16 DWORDs (64B)
          | (3 << 25);     // pad_amount:   code 3 -> 4 DWORDs (16B)
    g1[1] = (tile_k & 0xFFFF) << 16;                          // tensor_dim0[15:0] (== tile_k: no OOB)
    g1[2] = ((unsigned)tile_k >> 16) | ((tile_rows & 0xFFFF) << 16); // dim0 hi | tensor_dim1 lo
    g1[3] = ((unsigned)tile_rows >> 16) | ((tile_k & 0xFFFF) << 16); // dim1 hi | tile_dim0
    g1[4] = (tile_rows & 0xFFFF);                             // tile_dim1 | tile_dim2=0
    g1[5] = (int)(stride_elems & 0xFFFFFFFFll);               // tensor_dim0_stride[31:0]
    g1[6] = (int)((stride_elems >> 32) & 0xFFFFll);           // stride[47:32] | dim1_stride lo = 0
    g1[7] = 0;
    i32x4 z4 = {0, 0, 0, 0};
#if __has_builtin(__builtin_amdgcn_tensor_load_to_lds)
#if __clang_major__ >= 23
    i32x8 z8 = {0, 0, 0, 0, 0, 0, 0, 0};
    __builtin_amdgcn_tensor_load_to_lds(g0, g1, z4, z4, z8, 0);
#else
    __builtin_amdgcn_tensor_load_to_lds(g0, g1, z4, z4, 0);
#endif
#else
    (void)g0; (void)g1; (void)z4;
#endif
}

// ---------------- main GEMM: Y[row,col] = sum_k BigW[row,k]*Xbf[col,k], WMMA bf16
// grid = (75, 25), block = 256 (8 waves). WG tile: 64 rows x 256 cols. K loop: 50 x 32,
// double-buffered TDM (unrolled x2 so buffer indices are compile-time constants ->
// pure-SGPR descriptor setup): tile ii+1 DMAs while tile ii runs on the matrix pipes.
__global__ void __launch_bounds__(256)
gemm_kernel(const __bf16* __restrict__ BigW, const __bf16* __restrict__ Xbf,
            float* __restrict__ y_pre, float* __restrict__ stats) {
    __shared__ __align__(16) __bf16 Wl[2][64 * WL_STRIDE];    // 2 x  5.1 KB
    __shared__ __align__(16) __bf16 Xl[2][256 * WL_STRIDE];   // 2 x 20.5 KB
    __shared__ float lsum[64], lsq[64];

    const int rBase = blockIdx.y * 64;
    const int cBase = blockIdx.x * 256;
    const int tid  = threadIdx.x;
    const int lane = tid & 31;
    const int wave = tid >> 5;
    const int waveR = wave >> 2;   // 0..1 -> 32 rows each
    const int waveC = wave & 3;    // 0..3 -> 64 cols each
    const int mn    = lane & 15;
    const int khalf = lane >> 4;   // 0/1: which half of the lane group

    if (tid < 64) { lsum[tid] = 0.0f; lsq[tid] = 0.0f; }

    const __bf16* gW = BigW + (long long)rBase * KDIM;
    const __bf16* gX = Xbf  + (long long)cBase * KDIM;
    const unsigned wOff[2] = { (unsigned)(unsigned long long)(uintptr_t)(&Wl[0][0]),
                               (unsigned)(unsigned long long)(uintptr_t)(&Wl[1][0]) };
    const unsigned xOff[2] = { (unsigned)(unsigned long long)(uintptr_t)(&Xl[0][0]),
                               (unsigned)(unsigned long long)(uintptr_t)(&Xl[1][0]) };

    v8f acc[2][4] = {};

#if __has_builtin(__builtin_amdgcn_tensor_load_to_lds)
    if (wave == 0) {            // prologue: DMA tile 0 into buffer 0
        tdm_load_2d(wOff[0], gW, 32, 64,  KDIM);
        tdm_load_2d(xOff[0], gX, 32, 256, KDIM);
    }
#endif

    for (int i = 0; i < KSTEPS; i += 2) {
        #pragma unroll
        for (int half = 0; half < 2; ++half) {   // cur/nxt buffers become constants
            const int ii  = i + half;
            const int cur = half;
            const int nxt = half ^ 1;

            __syncthreads();   // all readers of buffer `nxt` (iteration ii-1) are done
#if __has_builtin(__builtin_amdgcn_tensor_load_to_lds)
            if (wave == 0) {
                if (ii + 1 < KSTEPS) {   // compile-time true for half==0
                    // prefetch next K tile into the alternate buffer (overlaps WMMA below)
                    tdm_load_2d(wOff[nxt], gW + (ii + 1) * 32, 32, 64,  KDIM);
                    tdm_load_2d(xOff[nxt], gX + (ii + 1) * 32, 32, 256, KDIM);
                    __builtin_amdgcn_s_wait_tensorcnt(2);  // in-order: current pair complete
                } else {
                    __builtin_amdgcn_s_wait_tensorcnt(0);
                }
            }
#else
            {   // fallback: manual staging through VGPRs into current buffer
                int kb = ii * 32;
                int row = tid >> 2, q = tid & 3;
                uint4 vv = *(const uint4*)(gW + (long long)row * KDIM + kb + q * 8);
                *(uint4*)(&Wl[cur][row * WL_STRIDE + q * 8]) = vv;
                const uint4* src = (const uint4*)(gX + (long long)tid * KDIM + kb);
                uint4 a0 = src[0], a1 = src[1], a2 = src[2], a3 = src[3];
                uint4* dst = (uint4*)(&Xl[cur][tid * WL_STRIDE]);
                dst[0] = a0; dst[1] = a1; dst[2] = a2; dst[3] = a3;
            }
#endif
            __syncthreads();   // current buffer published to all waves

            const __bf16* Wb = Wl[cur];
            const __bf16* Xb = Xl[cur];

            // A fragments (16x32 bf16, contiguous K per lane-half)
            v16bf afrag[2];
            #pragma unroll
            for (int rt = 0; rt < 2; ++rt) {
                const __bf16* p = &Wb[(waveR * 32 + rt * 16 + mn) * WL_STRIDE + khalf * 16];
                v8bf lo = *(const v8bf*)p;          // K = 16*khalf + 0..7
                v8bf hi = *(const v8bf*)(p + 8);    // K = 16*khalf + 8..15
                #pragma unroll
                for (int j = 0; j < 8; ++j) { afrag[rt][j] = lo[j]; afrag[rt][j + 8] = hi[j]; }
            }
            // B fragments (32x16 bf16, contiguous K per lane-half)
            v16bf bfrag[4];
            #pragma unroll
            for (int ct = 0; ct < 4; ++ct)
                bfrag[ct] = *(const v16bf*)(&Xb[(waveC * 64 + ct * 16 + mn) * WL_STRIDE + khalf * 16]);

            #pragma unroll
            for (int rt = 0; rt < 2; ++rt)
                #pragma unroll
                for (int ct = 0; ct < 4; ++ct)
                    acc[rt][ct] = __builtin_amdgcn_wmma_f32_16x16x32_bf16(
                        false, afrag[rt], false, bfrag[ct], (short)0, acc[rt][ct], false, false);
        }
    }

    // epilogue: scatter to y_pre[n,o,t,w] + per-channel BN partial sums.
    long long cb[4];
    #pragma unroll
    for (int ct = 0; ct < 4; ++ct) {
        int col = cBase + waveC * 64 + ct * 16 + mn;   // D layout: N = lane%16
        int n = col / TB, t = col - n * TB;
        cb[ct] = (long long)n * NTV + (long long)t * VB;
    }
    #pragma unroll
    for (int rt = 0; rt < 2; ++rt) {
        int base = rBase + waveR * 32 + rt * 16 + 8 * khalf;   // M = r + 8*(lane/16)
        int o = base / VB, w = base - o * VB;
        #pragma unroll
        for (int r = 0; r < 8; ++r) {
            long long ro = (long long)o * 7500 + w;            // o*(T*V) + w
            float s1 = 0.0f, s2 = 0.0f;
            #pragma unroll
            for (int ct = 0; ct < 4; ++ct) {
                float val = acc[rt][ct][r];
                y_pre[cb[ct] + ro] = val;
                s1 += val; s2 += val * val;
            }
            atomicAdd(&lsum[o], s1);    // same o across ct: pre-reduced 4x
            atomicAdd(&lsq[o],  s2);
            if (++w == VB) { w = 0; ++o; }
        }
    }
    __syncthreads();
    if (tid < 64) {
        atomicAdd(&stats[tid],      lsum[tid]);
        atomicAdd(&stats[64 + tid], lsq[tid]);
    }
}

// ---------------- BN finalize: scale/shift per channel ----------------------
__global__ void bnfin_kernel(float* stats, const float* __restrict__ gamma,
                             const float* __restrict__ beta) {
    int o = threadIdx.x;  // 64 threads
    float mean = stats[o] / (float)NTV;
    float var  = stats[64 + o] / (float)NTV - mean * mean;
    float sc   = gamma[o] * rsqrtf(var + 1e-5f);
    stats[128 + o] = sc;
    stats[192 + o] = beta[o] - mean * sc;
}

// ---------------- out = relu(y_pre*scale + shift + x), vectorized (ext-vector for NT builtins)
__global__ void final_kernel(const float* __restrict__ y_pre, const float* __restrict__ x,
                             const float* __restrict__ stats, float* __restrict__ out) {
    long long i4 = (long long)blockIdx.x * 256 + threadIdx.x;  // 7,680,000 = 30000*256
    long long idx = i4 * 4;
    int o = (int)((idx / 7500) % 64);   // 7500 = T*V, multiple of 4 -> o uniform in the quad
    float sc = stats[128 + o], sh = stats[192 + o];
    v4f y = __builtin_nontemporal_load((const v4f*)(y_pre + idx));   // streaming, last use
    v4f xr = *(const v4f*)(x + idx);
    v4f r;
    #pragma unroll
    for (int j = 0; j < 4; ++j)
        r[j] = fmaxf(y[j] * sc + sh + xr[j], 0.0f);
    __builtin_nontemporal_store(r, (v4f*)(out + idx));  // never re-read on device
}

extern "C" void kernel_launch(void* const* d_in, const int* in_sizes, int n_in,
                              void* d_out, int out_size, void* d_ws, size_t ws_size,
                              hipStream_t stream) {
    const float* x     = (const float*)d_in[0];
    const float* PA    = (const float*)d_in[1];
    const float* Wc    = (const float*)d_in[2];
    // d_in[3] = bc: per-channel constant bias is cancelled exactly by training-mode BN -> unused
    const float* gamma = (const float*)d_in[4];
    const float* beta  = (const float*)d_in[5];
    float* out = (float*)d_out;

    // workspace carve-up (all offsets 16B aligned)
    char* ws = (char*)d_ws;
    float*  stats = (float*)ws;                                   // 1 KB
    __bf16* BigW  = (__bf16*)(ws + 4096);                         // 5.12 MB
    __bf16* Xbf   = (__bf16*)(ws + 4096 + 5120000);               // 61.44 MB
    float*  y_pre = (float*)(ws + 4096 + 5120000 + 61440000);     // 122.88 MB

    zero_stats_kernel<<<1, 256, 0, stream>>>(stats);
    bigw_kernel<<<10000, 256, 0, stream>>>(PA, Wc, BigW);
    xprep_kernel<<<120000, 256, 0, stream>>>(x, Xbf);
    gemm_kernel<<<dim3(COLS / 256, ROWS / 64), 256, 0, stream>>>(BigW, Xbf, y_pre, stats);
    bnfin_kernel<<<1, 64, 0, stream>>>(stats, gamma, beta);
    final_kernel<<<30000, 256, 0, stream>>>(y_pre, x, stats, out);
}
